// MyMonotoneNN_dim2_36953898615396
// MI455X (gfx1250) — matrix operand order repaired and checked by reading, not compiled, not run on hardware
//
#include <hip/hip_runtime.h>

typedef __attribute__((ext_vector_type(16))) __bf16 v16bf;
typedef __attribute__((ext_vector_type(8)))  float  v8f;

#define LB   (-5.0f)
#define NM1  255

__device__ __forceinline__ float lrelu(float x) { return x > 0.f ? x : 0.01f * x; }

// ---------------------------------------------------------------------------
// Prep: convert W1 (128x128 f32, row-major, h0 @ W1) for both models into
// bf16 wave32 B-fragments for v_wmma_f32_16x16x32_bf16.
// B is KxN = 32x16 per tile: lanes 0-15 hold K=0..15 (element e -> K=e),
// lanes 16-31 hold K=16..31.
// Fragment (m, kb, nt): row k = 32*kb + e + (lane>=16 ? 16 : 0),
//                       col  = nt*16 + (lane & 15).
// One thread per (m, kb, nt, lane): 2*4*8*32 = 2048 threads, 16 elems each.
// ---------------------------------------------------------------------------
__global__ void prep_w1_frags(const float* __restrict__ w1_0,
                              const float* __restrict__ w1_1,
                              __bf16* __restrict__ frag) {
    int t    = blockIdx.x * blockDim.x + threadIdx.x;   // 0..2047
    int lane = t & 31;
    int nt   = (t >> 5) & 7;
    int kb   = (t >> 8) & 3;
    int m    = (t >> 10) & 1;
    const float* W = m ? w1_1 : w1_0;
    int col   = nt * 16 + (lane & 15);
    int kbase = 32 * kb + (lane >= 16 ? 16 : 0);
    __bf16* dst = frag + (size_t)(((m * 4 + kb) * 8 + nt) * 32 + lane) * 16;
#pragma unroll
    for (int e = 0; e < 16; ++e)
        dst[e] = (__bf16)W[(kbase + e) * 128 + col];
}

// ---------------------------------------------------------------------------
// Main: one block per (batch b, dim m). 128 threads = 4 waves, each wave does
// 4 tiles of 16 quadrature points. Per tile: build bf16 A-fragments from
// layer0 (rank-1 + LeakyReLU), 8 N-tiles x 4 K-steps of WMMA starting from
// inline-zero C (bias added post-GEMM), LeakyReLU, dot with W2 + cross-lane
// reduce, ELU + 1, accumulate over n.
//
// B-fragments are double-buffered across N-tiles so ds_load_b128 issues well
// ahead of the consuming WMMA chain (partial DScnt waits instead of
// s_wait_dscnt 0 right before every WMMA, as seen in round 2).
// ---------------------------------------------------------------------------
__global__ __launch_bounds__(128, 4)
void mono_mlp_kernel(const float* __restrict__ z, const float* __restrict__ u,
                     const float* __restrict__ w0a, const float* __restrict__ b0a,
                     const float* __restrict__ b1a, const float* __restrict__ w2a,
                     const float* __restrict__ b2a,
                     const float* __restrict__ w0b, const float* __restrict__ b0b,
                     const float* __restrict__ b1b, const float* __restrict__ w2b,
                     const float* __restrict__ b2b,
                     const __bf16* __restrict__ frag, float* __restrict__ hz) {
    __shared__ __attribute__((aligned(32))) __bf16 ldsFrag[16384];  // 32 KB W1 frags
    __shared__ __attribute__((aligned(8))) float lw0[128];
    __shared__ __attribute__((aligned(8))) float lb0[128];
    __shared__ float lb1[128], lw2[128];
    __shared__ float wsum[4];

    const int bm  = blockIdx.x;          // (b*2 + m)
    const int m   = bm & 1;
    const int tid = threadIdx.x;

    const float* w0 = m ? w0b : w0a;
    const float* b0 = m ? b0b : b0a;
    const float* b1 = m ? b1b : b1a;
    const float* w2 = m ? w2b : w2a;
    const float  b2 = m ? b2b[0] : b2a[0];

    // Stage small per-layer vectors and the 32 KB of W1 fragments into LDS.
    lw0[tid] = w0[tid];
    lb0[tid] = b0[tid];
    lb1[tid] = b1[tid];
    lw2[tid] = w2[tid];
    {
        const uint4* src = (const uint4*)(frag + (size_t)m * 16384);
        uint4*       dst = (uint4*)ldsFrag;
        for (int i = tid; i < 2048; i += 128) dst[i] = src[i];
    }
    __syncthreads();

    const float zc = fmaxf(z[bm], LB);
    const float d  = (zc - LB) * (1.0f / (float)NM1);
    const float* ub = u + (size_t)bm * NM1;

    const int lane    = tid & 31;
    const int wid     = tid >> 5;
    const int pm      = lane & 15;               // point-in-tile (M row)
    const int halfsel = (lane >= 16) ? 16 : 0;   // A-layout K offset for hi half

    // Per-lane bias / W2 entries for each N-tile (column = nt*16 + pm).
    float b1v[8], w2v[8];
#pragma unroll
    for (int nt = 0; nt < 8; ++nt) {
        b1v[nt] = lb1[nt * 16 + pm];
        w2v[nt] = lw2[nt * 16 + pm];
    }

    float acc = 0.f;  // sum over this wave's points of (elu(..)+1)

    // Uniform trip count: wid in {0..3}, exactly 4 iterations per wave,
    // so the in-loop __syncthreads() is legal for all threads.
    for (int t = wid; t < 16; t += 4) {          // 16 tiles of 16 points = N=256
        const int   n  = t * 16 + pm;
        const float uu = (n < NM1) ? ub[n] : 0.f;
        const float x  = fmaf(d, (float)n, LB) + uu * d;   // z_split + jitter

        // Kick off the B-fragment loads for nt=0 first; the ~200 VALU ops of
        // layer 0 below then cover the LDS round-trip latency.
        v16bf bcur[4], bnxt[4];
#pragma unroll
        for (int kb = 0; kb < 4; ++kb)
            bcur[kb] = *(const v16bf*)&ldsFrag[(size_t)((kb * 8) * 32 + lane) * 16];

        // Layer 0 -> A fragments (16-bit A 16x32 layout).
        // Element pairs (2e2, 2e2+1) map to consecutive hidden indices j, j+1,
        // j even -> 8B-aligned float2 LDS loads.
        v16bf afrag[4];
#pragma unroll
        for (int kb = 0; kb < 4; ++kb) {
#pragma unroll
            for (int e2 = 0; e2 < 8; ++e2) {
                int    e = 2 * e2;
                int    j = 32 * kb + e + halfsel;
                float2 w = *(const float2*)&lw0[j];
                float2 c = *(const float2*)&lb0[j];
                afrag[kb][e]     = (__bf16)lrelu(fmaf(x, w.x, c.x));
                afrag[kb][e + 1] = (__bf16)lrelu(fmaf(x, w.y, c.y));
            }
        }

        float tsum[8];
#pragma unroll
        for (int r = 0; r < 8; ++r) tsum[r] = 0.f;

#pragma unroll
        for (int nt = 0; nt < 8; ++nt) {
            // Prefetch next N-tile's fragments behind this tile's WMMA chain.
            if (nt < 7) {
#pragma unroll
                for (int kb = 0; kb < 4; ++kb)
                    bnxt[kb] = *(const v16bf*)&ldsFrag[(size_t)((kb * 8 + nt + 1) * 32 + lane) * 16];
            }
            // Inline-zero C for the first WMMA of the chain (SRC2 = 0);
            // layer-1 bias is added post-GEMM instead, freeing ~64 VGPRs of
            // broadcast C-inits seen in round 2.
            v8f c = {};
#pragma unroll
            for (int kb = 0; kb < 4; ++kb)
                c = __builtin_amdgcn_wmma_f32_16x16x32_bf16(
                        false, afrag[kb], false, bcur[kb], (short)0, c, false, false);
            // bias + LeakyReLU, then fold layer-2 column weight immediately.
#pragma unroll
            for (int r = 0; r < 8; ++r)
                tsum[r] = fmaf(lrelu(c[r] + b1v[nt]), w2v[nt], tsum[r]);
#pragma unroll
            for (int kb = 0; kb < 4; ++kb) bcur[kb] = bnxt[kb];  // rename only
        }

        // Reduce over the 16 lanes of each half (sum over hidden index j2).
#pragma unroll
        for (int mask = 1; mask <= 8; mask <<= 1) {
#pragma unroll
            for (int r = 0; r < 8; ++r)
                tsum[r] += __shfl_xor(tsum[r], mask, 32);
        }
        // ELU + 1 per point; each half owns 8 points (M = r + 8*(lane >= 16)).
        float hsumv = 0.f;
#pragma unroll
        for (int r = 0; r < 8; ++r) {
            float s = tsum[r] + b2;
            float y = s > 0.f ? s : (__expf(s) - 1.f);
            hsumv += y + 1.f;
        }
        acc += hsumv;

        // Barrier = LDS clobber: keeps fragments / w0 / b0 from being hoisted
        // across tiles into registers (uniform control flow).
        __syncthreads();
    }

    acc += __shfl_xor(acc, 16, 32);              // combine the two halves
    if (lane == 0) wsum[wid] = acc;
    __syncthreads();
    if (tid == 0)
        hz[bm] = d * (wsum[0] + wsum[1] + wsum[2] + wsum[3]);
}

// ---------------------------------------------------------------------------
// Finalize: out[b] = Hz[b,0]*lin_w[0] + Hz[b,1]*lin_w[1] + lin_b
// ---------------------------------------------------------------------------
__global__ void finalize_kernel(const float* __restrict__ hz,
                                const float* __restrict__ lin_w,
                                const float* __restrict__ lin_b,
                                float* __restrict__ out, int nb) {
    int b = blockIdx.x * blockDim.x + threadIdx.x;
    if (b < nb)
        out[b] = hz[b * 2] * lin_w[0] + hz[b * 2 + 1] * lin_w[1] + lin_b[0];
}

extern "C" void kernel_launch(void* const* d_in, const int* in_sizes, int n_in,
                              void* d_out, int out_size, void* d_ws, size_t ws_size,
                              hipStream_t stream) {
    const float* z     = (const float*)d_in[0];
    const float* u     = (const float*)d_in[1];
    // d_in[2] = N (int scalar) -- fixed at 256 by the reference setup
    const float* z0_w0 = (const float*)d_in[3];
    const float* z0_b0 = (const float*)d_in[4];
    const float* z0_w1 = (const float*)d_in[5];
    const float* z0_b1 = (const float*)d_in[6];
    const float* z0_w2 = (const float*)d_in[7];
    const float* z0_b2 = (const float*)d_in[8];
    const float* z1_w0 = (const float*)d_in[9];
    const float* z1_b0 = (const float*)d_in[10];
    const float* z1_w1 = (const float*)d_in[11];
    const float* z1_b1 = (const float*)d_in[12];
    const float* z1_w2 = (const float*)d_in[13];
    const float* z1_b2 = (const float*)d_in[14];
    const float* lin_w = (const float*)d_in[15];
    const float* lin_b = (const float*)d_in[16];

    const int nb  = in_sizes[0] / 2;  // 2048 batch rows
    const int nbm = nb * 2;           // 4096 (b, m) pairs

    __bf16* fragp = (__bf16*)d_ws;                      // 64 KB: W1 bf16 fragments
    float*  hzp   = (float*)((char*)d_ws + 65536);      // 16 KB: Hz[b,m]

    prep_w1_frags<<<8, 256, 0, stream>>>(z0_w1, z1_w1, fragp);

    mono_mlp_kernel<<<nbm, 128, 0, stream>>>(
        z, u,
        z0_w0, z0_b0, z0_b1, z0_w2, z0_b2,
        z1_w0, z1_b0, z1_b1, z1_w2, z1_b2,
        fragp, hzp);

    finalize_kernel<<<(nb + 255) / 256, 256, 0, stream>>>(hzp, lin_w, lin_b,
                                                          (float*)d_out, nb);
}
